// SparKEncoder_75797582840358
// MI455X (gfx1250) — matrix operand order repaired and checked by reading, not compiled, not run on hardware
//
#include <hip/hip_runtime.h>
#include <hip/hip_bf16.h>

// ---------------------------------------------------------------------------
// SparK-style sparse 3D conv encoder for MI455X (gfx1250, wave32, WMMA).
//
// All matmuls run on v_wmma_f32_16x16x32_f16 (f16 A/B, f32 accumulate).
// Conv GEMM register-blocks 4 M-tiles per wave so each B (weight) fragment
// feeds 4 WMMAs: 4x less weight traffic (critical for stage3, whose 255MB
// w2 exceeds the 192MB L2) and 4x arithmetic intensity per wave.
// Kernel maps (grid scatter, 27-neighbor lookup, stride-2 unique-downsample)
// are computed on device; the downsample compaction is a deterministic
// ordered prefix scan that reproduces np.unique's lexicographic order.
//
// Input flattening assumption (compile-only environment): d_in =
//   [0] dense_features f32 (2*768*32^3), [1] unmasked_coords i32 (N*4),
//   [2] proj_w, [3] proj_b, then per stage s: [4+6s..] w1,g1,b1,w2,g2,b2.
// ---------------------------------------------------------------------------

typedef __attribute__((ext_vector_type(16))) _Float16 v16h;
typedef __attribute__((ext_vector_type(8)))  float    v8f;

#define SPK_EPS   1e-5f
#define SPK_D     32
#define SPK_DLIN  32768      // 32^3
#define SPK_CELLS 65536      // B(2) * 32^3
#define SPK_CMAX  1536
#define SPK_MT    4          // M-tiles register-blocked per wave in conv GEMM

// ---------------------------------------------------------------- utilities

__global__ void k_init_counts(int* counts, int n0) {
  if (threadIdx.x == 0 && blockIdx.x == 0) counts[0] = n0;
}

__global__ void k_fill_i32(int* __restrict__ g, int val, int n) {
  int i = blockIdx.x * blockDim.x + threadIdx.x;
  if (i < n) g[i] = val;
}

// grid[b,x,y,z] = row index
__global__ void k_scatter_grid(int* __restrict__ g,
                               const int* __restrict__ coords,
                               const int* __restrict__ counts, int cidx) {
  int i = blockIdx.x * blockDim.x + threadIdx.x;
  if (i >= counts[cidx]) return;
  int b = coords[4 * i + 0], x = coords[4 * i + 1];
  int y = coords[4 * i + 2], z = coords[4 * i + 3];
  g[b * SPK_DLIN + x * 1024 + y * 32 + z] = i;
}

// nbr[k, i] = input row hit by offset k (or -1), offsets scaled by ts.
// _OFFS order: dx outer, dy mid, dz inner.
__global__ void k_build_nbr(int* __restrict__ nbr,
                            const int* __restrict__ grid,
                            const int* __restrict__ coords,
                            const int* __restrict__ counts, int cidx,
                            int ts, int Mcap) {
  int i = blockIdx.x * blockDim.x + threadIdx.x;
  if (i >= counts[cidx] || i >= Mcap) return;
  int b = coords[4 * i + 0], x = coords[4 * i + 1];
  int y = coords[4 * i + 2], z = coords[4 * i + 3];
#pragma unroll
  for (int k = 0; k < 27; ++k) {
    int dx = k / 9 - 1, dy = (k / 3) % 3 - 1, dz = k % 3 - 1;
    int cx = x + dx * ts, cy = y + dy * ts, cz = z + dz * ts;
    int idx = -1;
    if (cx >= 0 && cx < SPK_D && cy >= 0 && cy < SPK_D && cz >= 0 && cz < SPK_D)
      idx = grid[b * SPK_DLIN + cx * 1024 + cy * 32 + cz];
    nbr[k * Mcap + i] = idx;
  }
}

// mark occupied stride-aligned cells (idempotent stores -> deterministic)
__global__ void k_mark_cells(int* __restrict__ g,
                             const int* __restrict__ coords,
                             const int* __restrict__ counts, int cidx,
                             int ts_out) {
  int i = blockIdx.x * blockDim.x + threadIdx.x;
  if (i >= counts[cidx]) return;
  int b = coords[4 * i + 0];
  int x = coords[4 * i + 1] / ts_out * ts_out;
  int y = coords[4 * i + 2] / ts_out * ts_out;
  int z = coords[4 * i + 3] / ts_out * ts_out;
  g[b * SPK_DLIN + x * 1024 + y * 32 + z] = 1;
}

// Single-workgroup ordered compaction over all 65536 cells. Linear cell
// index == lexicographic (b,x,y,z) => matches np.unique ordering. 100%
// deterministic (fixed scan association, no atomics).
__global__ __launch_bounds__(1024) void k_compact_cells(
    const int* __restrict__ mark, int* __restrict__ out_coords,
    int* __restrict__ counts, int out_idx) {
  __shared__ int sc[1024];
  __shared__ int s_base;
  int t = threadIdx.x;
  if (t == 0) s_base = 0;
  __syncthreads();
  for (int chunk = 0; chunk < SPK_CELLS; chunk += 1024) {
    int idx = chunk + t;
    int f = mark[idx] ? 1 : 0;
    sc[t] = f;
    __syncthreads();
    for (int o = 1; o < 1024; o <<= 1) {          // Hillis-Steele inclusive scan
      int v = (t >= o) ? sc[t - o] : 0;
      __syncthreads();
      sc[t] += v;
      __syncthreads();
    }
    if (f) {
      int pos = s_base + sc[t] - 1;
      int b = idx >> 15, lin = idx & (SPK_DLIN - 1);
      out_coords[4 * pos + 0] = b;
      out_coords[4 * pos + 1] = lin >> 10;
      out_coords[4 * pos + 2] = (lin >> 5) & 31;
      out_coords[4 * pos + 3] = lin & 31;
    }
    __syncthreads();
    if (t == 1023) s_base += sc[1023];
    __syncthreads();
  }
  if (t == 0) counts[out_idx] = s_base;
}

// ------------------------------------------------------------- WMMA GEMMs
// Fragment layouts per CDNA5 ISA 7.12.2 (wave32):
//  A (16x32 f16): half = lane>>4 ; row M = lane&15 ;
//     elems 0..7  -> K = kc + 8*half + e ; elems 8..15 -> K = kc+16+8*half+(e-8)
//  B (32x16 f16): col N = lane&15 ; same K mapping as A
//  C/D (16x16 f32): N = lane&15 ; elem v -> M = 8*half + v

// Projection: X[m, 0:96] = dense[b, :, x,y,z] @ proj_w + proj_b
// A is gathered with channel stride 32^3 from the dense tensor.
__global__ __launch_bounds__(32) void k_proj_gemm(
    const float* __restrict__ dense, const int* __restrict__ coords,
    const int* __restrict__ counts, const float* __restrict__ Wp,
    const float* __restrict__ bias, float* __restrict__ X) {
  const int M = counts[0];
  const int tm = blockIdx.x, tn = blockIdx.y;
  if (tm * 16 >= M) return;                       // uniform per wave: EXEC stays full
  const int lane = threadIdx.x, half = lane >> 4, l16 = lane & 15;
  const int mrow = tm * 16 + l16;
  const bool avalid = (mrow < M);
  long abase = 0;
  if (avalid) {
    int b = coords[4 * mrow + 0], x = coords[4 * mrow + 1];
    int y = coords[4 * mrow + 2], z = coords[4 * mrow + 3];
    abase = (long)b * 768 * SPK_DLIN + x * 1024 + y * 32 + z;
  }
  v8f acc;
#pragma unroll
  for (int v = 0; v < 8; ++v) acc[v] = 0.0f;

  for (int kc = 0; kc < 768; kc += 32) {
    v16h a;
#pragma unroll
    for (int e = 0; e < 16; ++e) a[e] = (_Float16)0.0f;
    if (avalid) {
      const int k0 = kc + half * 8;
#pragma unroll
      for (int e = 0; e < 8; ++e) {
        a[e]     = (_Float16)dense[abase + (long)(k0 + e) * SPK_DLIN];
        a[e + 8] = (_Float16)dense[abase + (long)(k0 + 16 + e) * SPK_DLIN];
      }
    }
    v16h bb;
    const int kb = kc + half * 8;
#pragma unroll
    for (int e = 0; e < 16; ++e)
      bb[e] = (_Float16)Wp[(long)(kb + e + (e & 8)) * 96 + tn * 16 + l16];

    acc = __builtin_amdgcn_wmma_f32_16x16x32_f16(false, a, false, bb,
                                                 (short)0, acc, false, false);
  }
#pragma unroll
  for (int v = 0; v < 8; ++v) {
    int m = tm * 16 + half * 8 + v;
    if (m < M) X[(long)m * 96 + tn * 16 + l16] = acc[v] + bias[tn * 16 + l16];
  }
}

// Sparse conv as 27-offset gather-GEMM: Y[m,:] = sum_k X[nbr[k,m],:] @ W[k].
// SPK_MT m-tiles per wave: each B (weight) fragment is loaded once and feeds
// SPK_MT WMMAs against SPK_MT gathered A fragments / accumulators.
__global__ __launch_bounds__(32) void k_conv_gemm(
    const float* __restrict__ X, const float* __restrict__ W,
    const int* __restrict__ nbr, float* __restrict__ Y,
    const int* __restrict__ counts, int cidx, int Cin, int Cout, int Mcap) {
  const int M = counts[cidx];
  const int tm0 = blockIdx.x * SPK_MT;            // first m-tile of this wave
  const int tn = blockIdx.y;
  if (tm0 * 16 >= M) return;                      // uniform -> EXEC full for WMMA
  const int lane = threadIdx.x, half = lane >> 4, l16 = lane & 15;

  v8f acc[SPK_MT];
#pragma unroll
  for (int t = 0; t < SPK_MT; ++t)
#pragma unroll
    for (int v = 0; v < 8; ++v) acc[t][v] = 0.0f;

  for (int ko = 0; ko < 27; ++ko) {
    // gather row index per m-tile for this kernel offset
    int  arow[SPK_MT];
    bool avalid[SPK_MT];
#pragma unroll
    for (int t = 0; t < SPK_MT; ++t) {
      const int mrow = tm0 * 16 + t * 16 + l16;
      const int r = (mrow < Mcap) ? nbr[ko * Mcap + mrow] : -1;
      avalid[t] = (mrow < M) && (r >= 0);
      arow[t] = avalid[t] ? r : 0;
    }
    const float* wp = W + (long)ko * Cin * Cout + tn * 16;

    for (int kc = 0; kc < Cin; kc += 32) {
      // B fragment: loaded ONCE, reused by SPK_MT WMMAs
      v16h bb;
      const int kb = kc + half * 8;
#pragma unroll
      for (int e = 0; e < 16; ++e)
        bb[e] = (_Float16)wp[(long)(kb + e + (e & 8)) * Cout + l16];

#pragma unroll
      for (int t = 0; t < SPK_MT; ++t) {
        v16h a;
#pragma unroll
        for (int e = 0; e < 16; ++e) a[e] = (_Float16)0.0f;
        if (avalid[t]) {
          const float* arp = X + (long)arow[t] * Cin;
          const int k0 = kc + half * 8;
          const float4 p0 = *(const float4*)(arp + k0);
          const float4 p1 = *(const float4*)(arp + k0 + 4);
          const float4 p2 = *(const float4*)(arp + k0 + 16);
          const float4 p3 = *(const float4*)(arp + k0 + 20);
          a[0] = (_Float16)p0.x;  a[1] = (_Float16)p0.y;
          a[2] = (_Float16)p0.z;  a[3] = (_Float16)p0.w;
          a[4] = (_Float16)p1.x;  a[5] = (_Float16)p1.y;
          a[6] = (_Float16)p1.z;  a[7] = (_Float16)p1.w;
          a[8]  = (_Float16)p2.x; a[9]  = (_Float16)p2.y;
          a[10] = (_Float16)p2.z; a[11] = (_Float16)p2.w;
          a[12] = (_Float16)p3.x; a[13] = (_Float16)p3.y;
          a[14] = (_Float16)p3.z; a[15] = (_Float16)p3.w;
        }
        acc[t] = __builtin_amdgcn_wmma_f32_16x16x32_f16(false, a, false, bb,
                                                        (short)0, acc[t],
                                                        false, false);
      }
    }
  }
#pragma unroll
  for (int t = 0; t < SPK_MT; ++t)
#pragma unroll
    for (int v = 0; v < 8; ++v) {
      int m = (tm0 + t) * 16 + half * 8 + v;
      if (m < M) Y[(long)m * Cout + tn * 16 + l16] = acc[t][v];
    }
}

// ---------------------------------------------------------------- BN + ReLU

// One block per channel, deterministic fixed-tree reduction (no float atomics)
__global__ __launch_bounds__(256) void k_bn_stats(
    const float* __restrict__ Y, const int* __restrict__ counts, int cidx,
    float* __restrict__ stats, int Cout) {
  __shared__ float ss[256], sq[256];
  const int M = counts[cidx];
  const int c = blockIdx.x, t = threadIdx.x;
  float s = 0.0f, q = 0.0f;
  for (int r = t; r < M; r += 256) {
    float v = Y[(long)r * Cout + c];
    s += v;
    q += v * v;
  }
  ss[t] = s; sq[t] = q;
  __syncthreads();
  for (int o = 128; o > 0; o >>= 1) {
    if (t < o) { ss[t] += ss[t + o]; sq[t] += sq[t + o]; }
    __syncthreads();
  }
  if (t == 0) { stats[c] = ss[0]; stats[SPK_CMAX + c] = sq[0]; }
}

__global__ void k_bn_apply(const float* __restrict__ Y,
                           const float* __restrict__ stats,
                           const float* __restrict__ gamma,
                           const float* __restrict__ beta,
                           const int* __restrict__ counts, int cidx,
                           float* __restrict__ dst, int Cout) {
  const int M = counts[cidx];
  long i = (long)blockIdx.x * blockDim.x + threadIdx.x;
  if (i >= (long)M * Cout) return;
  int c = (int)(i % Cout);
  float inv = 1.0f / (float)M;
  float mu  = stats[c] * inv;
  float var = stats[SPK_CMAX + c] * inv - mu * mu;
  float v = (Y[i] - mu) * rsqrtf(var + SPK_EPS) * gamma[c] + beta[c];
  dst[i] = v > 0.0f ? v : 0.0f;
}

// --------------------------------------------------------- final flat pack
// out = [x_final | inter0 | inter1 | inter2 | inter3], sizes from device counts
__global__ void k_pack(float* __restrict__ out,
                       const float* __restrict__ i0, const float* __restrict__ i1,
                       const float* __restrict__ i2, const float* __restrict__ i3,
                       const int* __restrict__ counts) {
  int tid = blockIdx.x * blockDim.x + threadIdx.x;
  int n1 = counts[1], n2 = counts[2], n3 = counts[3];
  int s0 = n3 * 1536, s1 = n1 * 192, s2 = n2 * 384, s3 = n3 * 768, s4 = n3 * 1536;
  int t = tid;
  if (t < s0) { out[tid] = i3[t]; return; } t -= s0;
  if (t < s1) { out[tid] = i0[t]; return; } t -= s1;
  if (t < s2) { out[tid] = i1[t]; return; } t -= s2;
  if (t < s3) { out[tid] = i2[t]; return; } t -= s3;
  if (t < s4) { out[tid] = i3[t]; }
}

// ---------------------------------------------------------------------------

extern "C" void kernel_launch(void* const* d_in, const int* in_sizes, int n_in,
                              void* d_out, int out_size, void* d_ws, size_t ws_size,
                              hipStream_t stream) {
  (void)n_in; (void)out_size; (void)ws_size;
  const float* dense   = (const float*)d_in[0];
  const int*   coords0 = (const int*)d_in[1];
  const float* proj_w  = (const float*)d_in[2];
  const float* proj_b  = (const float*)d_in[3];
  const float *W1[4], *G1[4], *B1[4], *W2[4], *G2[4], *B2[4];
  for (int s = 0; s < 4; ++s) {
    W1[s] = (const float*)d_in[4 + 6 * s + 0];
    G1[s] = (const float*)d_in[4 + 6 * s + 1];
    B1[s] = (const float*)d_in[4 + 6 * s + 2];
    W2[s] = (const float*)d_in[4 + 6 * s + 3];
    G2[s] = (const float*)d_in[4 + 6 * s + 4];
    B2[s] = (const float*)d_in[4 + 6 * s + 5];
  }
  const int N0 = in_sizes[1] / 4;

  // Static caps on downsampled point counts: N1<=2*16^3, N2<=2*8^3, N3<=2*4^3
  const int CAP1 = 8192, CAP2 = 1024, CAP3 = 128;
  const int caps[5] = {N0, CAP1, CAP2, CAP3, CAP3};
  const int tss[4]  = {1, 2, 4, 8};

  // ---- carve workspace (~53 MB) -------------------------------------------
  char* ws = (char*)d_ws;
  size_t off = 0;
  auto carve = [&](size_t bytes) -> char* {
    off = (off + 255) & ~(size_t)255;
    char* p = ws + off;
    off += bytes;
    return p;
  };
  int*   counts  = (int*)carve(32 * sizeof(int));
  int*   grid_a  = (int*)carve((size_t)SPK_CELLS * sizeof(int));
  int*   grid_b  = (int*)carve((size_t)SPK_CELLS * sizeof(int));
  int*   coords1 = (int*)carve((size_t)4 * CAP1 * sizeof(int));
  int*   coords2 = (int*)carve((size_t)4 * CAP2 * sizeof(int));
  int*   coords3 = (int*)carve((size_t)4 * CAP3 * sizeof(int));
  int*   nbr     = (int*)carve((size_t)27 * N0 * sizeof(int));
  float* Xbuf    = (float*)carve((size_t)N0 * 192 * sizeof(float));
  float* Ybuf    = (float*)carve((size_t)N0 * 192 * sizeof(float));
  float* i0buf   = (float*)carve((size_t)CAP1 * 192 * sizeof(float));
  float* i1buf   = (float*)carve((size_t)CAP2 * 384 * sizeof(float));
  float* i2buf   = (float*)carve((size_t)CAP3 * 768 * sizeof(float));
  float* i3buf   = (float*)carve((size_t)CAP3 * 1536 * sizeof(float));
  float* stats   = (float*)carve((size_t)2 * SPK_CMAX * sizeof(float));
  float* interbuf[4] = {i0buf, i1buf, i2buf, i3buf};
  int*   nextC[4]    = {coords1, coords2, coords3, nullptr};

  // ---- projection: x0 = gather(dense) @ proj_w + proj_b -------------------
  k_init_counts<<<1, 64, 0, stream>>>(counts, N0);
  {
    dim3 g((N0 + 15) / 16, 96 / 16);
    k_proj_gemm<<<g, 32, 0, stream>>>(dense, coords0, counts, proj_w, proj_b, Xbuf);
  }

  // ---- stages -------------------------------------------------------------
  const float* Xin = Xbuf;
  const int* curCoords = coords0;
  const int MB = 16 * SPK_MT;                     // rows per conv GEMM block
  for (int s = 0; s < 4; ++s) {
    const int ch = 96 << s, c2 = 2 * ch;
    const int capIn = caps[s], capOut = caps[s + 1];
    const int ts = tss[s];

    // conv1 (stride 1) kernel map
    k_fill_i32<<<SPK_CELLS / 256, 256, 0, stream>>>(grid_a, -1, SPK_CELLS);
    k_scatter_grid<<<(capIn + 255) / 256, 256, 0, stream>>>(grid_a, curCoords, counts, s);
    k_build_nbr<<<(capIn + 255) / 256, 256, 0, stream>>>(nbr, grid_a, curCoords,
                                                         counts, s, ts, capIn);
    // conv1 -> Y, BN+ReLU -> Xbuf
    {
      dim3 g((capIn + MB - 1) / MB, c2 / 16);
      k_conv_gemm<<<g, 32, 0, stream>>>(Xin, W1[s], nbr, Ybuf, counts, s, ch, c2, capIn);
    }
    k_bn_stats<<<c2, 256, 0, stream>>>(Ybuf, counts, s, stats, c2);
    k_bn_apply<<<(unsigned)(((long)capIn * c2 + 255) / 256), 256, 0, stream>>>(
        Ybuf, stats, G1[s], B1[s], counts, s, Xbuf, c2);

    // downsample (stride 2 except last stage)
    const int* outCoords;
    int oidx;
    if (s < 3) {
      k_fill_i32<<<SPK_CELLS / 256, 256, 0, stream>>>(grid_b, 0, SPK_CELLS);
      k_mark_cells<<<(capIn + 255) / 256, 256, 0, stream>>>(grid_b, curCoords,
                                                            counts, s, ts * 2);
      k_compact_cells<<<1, 1024, 0, stream>>>(grid_b, nextC[s], counts, s + 1);
      outCoords = nextC[s];
      oidx = s + 1;
    } else {
      outCoords = curCoords;
      oidx = s;
    }

    // conv2 kernel map uses the (still intact) input-coordinate grid_a
    k_build_nbr<<<(capOut + 255) / 256, 256, 0, stream>>>(nbr, grid_a, outCoords,
                                                          counts, oidx, ts, capOut);
    {
      dim3 g((capOut + MB - 1) / MB, c2 / 16);
      k_conv_gemm<<<g, 32, 0, stream>>>(Xbuf, W2[s], nbr, Ybuf, counts, oidx, c2, c2, capOut);
    }
    k_bn_stats<<<c2, 256, 0, stream>>>(Ybuf, counts, oidx, stats, c2);
    k_bn_apply<<<(unsigned)(((long)capOut * c2 + 255) / 256), 256, 0, stream>>>(
        Ybuf, stats, G2[s], B2[s], counts, oidx, interbuf[s], c2);

    Xin = interbuf[s];
    curCoords = outCoords;
  }

  // ---- pack flat tuple output ---------------------------------------------
  const int capTotal = CAP3 * 1536 + CAP1 * 192 + CAP2 * 384 + CAP3 * 768 + CAP3 * 1536;
  k_pack<<<(capTotal + 255) / 256, 256, 0, stream>>>((float*)d_out, i0buf, i1buf,
                                                     i2buf, i3buf, counts);
}
